// GatedLinearAttention_13632226197673
// MI455X (gfx1250) — compile-verified, matrix operand order checked by reading, not run on hardware
//
#include <hip/hip_runtime.h>

// Problem constants (match reference)
#define B_   2
#define T_   2048
#define C_   1024
#define H_   16
#define DK_  64
#define DV_  64
#define CONVK 4
#define EPS_ 1e-5f
#define M_   (B_ * T_)          // 4096 rows

typedef __attribute__((ext_vector_type(16))) __bf16 v16bf;
typedef __attribute__((ext_vector_type(8)))  __bf16 v8bf;
typedef __attribute__((ext_vector_type(8)))  float  v8f;
typedef unsigned int v4ui __attribute__((ext_vector_type(4)));
typedef int          v8i  __attribute__((ext_vector_type(8)));
typedef int          v4i  __attribute__((ext_vector_type(4)));

__device__ __forceinline__ float sigf(float x)  { return 1.f / (1.f + __expf(-x)); }
__device__ __forceinline__ float siluf(float x) { return x * sigf(x); }

// ---------------------------------------------------------------------------
// f32 -> bf16 elementwise convert
// ---------------------------------------------------------------------------
__global__ __launch_bounds__(256) void cvt_bf16_kernel(const float* __restrict__ x,
                                                       __bf16* __restrict__ y, int n) {
    int i = blockIdx.x * 256 + threadIdx.x;
    if (i < n) y[i] = (__bf16)x[i];
}

// ---------------------------------------------------------------------------
// Pack a row-major [K, N] f32 weight into the WMMA B-operand lane layout:
// Bp[((nt*ktiles + kt)*32 + lane)*16 + i] = W[(kt*32 + kk), nt*16 + nn]
//   nn = lane & 15
//   lane < 16 : kk = (i<8) ? i      : i+8    (K 0-7, 16-23)
//   lane >= 16: kk = (i<8) ? i+8    : i+16   (K 8-15, 24-31)
// ---------------------------------------------------------------------------
__global__ __launch_bounds__(256) void pack_weight_kernel(const float* __restrict__ W,
                                                          __bf16* __restrict__ Bp,
                                                          int K, int Ntot) {
    size_t id = (size_t)blockIdx.x * 256 + threadIdx.x;
    size_t total = (size_t)K * Ntot;
    if (id >= total) return;
    int i    = (int)(id & 15);
    int lane = (int)((id >> 4) & 31);
    int ktiles = K >> 5;
    int kt = (int)((id >> 9) % (size_t)ktiles);
    int nt = (int)((id >> 9) / (size_t)ktiles);
    int kk = (lane < 16) ? ((i < 8) ? i : i + 8) : ((i < 8) ? i + 8 : i + 16);
    int nn = lane & 15;
    Bp[id] = (__bf16)W[(size_t)(kt * 32 + kk) * Ntot + nt * 16 + nn];
}

// ---------------------------------------------------------------------------
// TDM: issue one 2D tensor_load_to_lds that pulls a 64(N) x 32(K) packed
// B panel (4 n-tiles, each 512 contiguous bf16 = 1KB row, row stride =
// ktiles*512 elements) into LDS at byte offset lds_off.
// D# built per CDNA5 ISA.
// ---------------------------------------------------------------------------
__device__ __forceinline__ void tdm_load_bpanel(const __bf16* gtile, unsigned lds_off,
                                                int ktiles, int ntiles) {
    unsigned long long ga = (unsigned long long)(uintptr_t)gtile;
    const unsigned rowElems    = 512u;                      // 32 lanes * 16 bf16
    const unsigned strideElems = (unsigned)ktiles * 512u;   // row stride (elements)

    v4ui g0;
    g0.x = 1u;                                              // count=1 (valid user D#)
    g0.y = lds_off;                                         // LDS byte address
    g0.z = (unsigned)(ga & 0xFFFFFFFFu);                    // global_addr[31:0]
    g0.w = (unsigned)((ga >> 32) & 0x01FFFFFFu)             // global_addr[56:32]
         | (2u << 30);                                      // type = 2 ("image")

    v8i g1;
    g1[0] = (int)(1u << 16);                                // data_size=1 (2 bytes)
    g1[1] = (int)((rowElems & 0xFFFFu) << 16);              // tensor_dim0[15:0]
    g1[2] = (int)((rowElems >> 16) |
                  (((unsigned)ntiles & 0xFFFFu) << 16));    // tensor_dim0 hi | tensor_dim1 lo
    g1[3] = (int)((((unsigned)ntiles >> 16) & 0xFFFFu) |
                  ((rowElems & 0xFFFFu) << 16));            // tensor_dim1 hi | tile_dim0
    g1[4] = (int)4u;                                        // tile_dim1=4, tile_dim2=0
    g1[5] = (int)strideElems;                               // tensor_dim0_stride[31:0]
    g1[6] = 0;                                              // stride hi | dim1_stride lo
    g1[7] = 0;

    v4i gz = (v4i){0, 0, 0, 0};                             // 2D tensor: groups 2/3 zero
#if __clang_major__ >= 23
    v8i gz8 = (v8i){0, 0, 0, 0, 0, 0, 0, 0};
    __builtin_amdgcn_tensor_load_to_lds(g0, g1, gz, gz, gz8, 0);
#else
    __builtin_amdgcn_tensor_load_to_lds(g0, g1, gz, gz, 0);
#endif
}

// ---------------------------------------------------------------------------
// bf16 WMMA GEMM: C[M,N] = A[M,K] * Bpacked[K,N]  (f32 accumulate)
// block = 256 threads (8 waves); block tile = 128(M) x 64(N).
// B panels are staged into LDS by the Tensor Data Mover (double-buffered,
// wave 0 issues + waits TENSORcnt); all 8 waves consume the panel from LDS.
// Each wave: one 16-row strip x 4 N-tiles -> 4 accumulators, K/32 wmma steps.
// ---------------------------------------------------------------------------
__global__ __launch_bounds__(256) void gemm_bf16_wmma(const __bf16* __restrict__ A,
                                                      const __bf16* __restrict__ Bp,
                                                      float* __restrict__ Cmat,
                                                      int M, int N, int K) {
    __shared__ __align__(32) __bf16 bsm[2][2048];           // 2 x 4KB B panels

    const int lane = threadIdx.x & 31;
    const int wave = threadIdx.x >> 5;           // 0..7
    const int m0   = blockIdx.y * 128 + wave * 16;
    const int n0   = blockIdx.x * 64;
    const int half = lane >> 4;                  // 0 or 1
    const int row  = m0 + (lane & 15);
    const int ktiles = K >> 5;
    const int ntiles = N >> 4;
    const int nt0    = n0 >> 4;

    v8f acc[4];
#pragma unroll
    for (int j = 0; j < 4; ++j) acc[j] = (v8f){0.f,0.f,0.f,0.f,0.f,0.f,0.f,0.f};

    // LDS flat addresses: low 32 bits of the generic address are the LDS offset
    const unsigned lds0 = (unsigned)(uintptr_t)&bsm[0][0];
    const unsigned lds1 = (unsigned)(uintptr_t)&bsm[1][0];

    if (wave == 0) {    // prologue: panel kt=0 -> buffer 0
        tdm_load_bpanel(Bp + (size_t)nt0 * ktiles * 512, lds0, ktiles, ntiles);
    }

    const __bf16* arow = A + (size_t)row * K + half * 8;

    for (int kt = 0; kt < ktiles; ++kt) {
        if (wave == 0) {
            if (kt + 1 < ktiles) {
                tdm_load_bpanel(Bp + ((size_t)nt0 * ktiles + kt + 1) * 512,
                                ((kt + 1) & 1) ? lds1 : lds0, ktiles, ntiles);
                __builtin_amdgcn_s_wait_tensorcnt((short)1);   // panel kt landed
            } else {
                __builtin_amdgcn_s_wait_tensorcnt((short)0);
            }
        }
        __syncthreads();                       // publish panel kt to all waves

        if (kt + 1 < ktiles)
            __builtin_prefetch(arow + (size_t)(kt + 1) * 32, 0, 0);

        // A operand: 16-bit 16x32 layout, two aligned 16B chunks per lane
        v8bf a0 = *(const v8bf*)(arow + (size_t)kt * 32);
        v8bf a1 = *(const v8bf*)(arow + (size_t)kt * 32 + 16);
        v16bf av = __builtin_shufflevector(a0, a1,
            0,1,2,3,4,5,6,7,8,9,10,11,12,13,14,15);

        const __bf16* bpan = bsm[kt & 1];
#pragma unroll
        for (int j = 0; j < 4; ++j) {
            const v16bf bv = *(const v16bf*)(bpan + ((size_t)j * 32 + lane) * 16);
            acc[j] = __builtin_amdgcn_wmma_f32_16x16x32_bf16(
                false, av, false, bv, (short)0, acc[j], false, false);
        }
        __syncthreads();                       // all reads done before buffer reuse
    }

    // C layout: VGPR r -> M = m0 + half*8 + r, N = n0 + j*16 + (lane&15)
#pragma unroll
    for (int j = 0; j < 4; ++j) {
        int col = n0 + j * 16 + (lane & 15);
        float* cp = Cmat + (size_t)(m0 + half * 8) * N + col;
#pragma unroll
        for (int r = 0; r < 8; ++r) cp[(size_t)r * N] = acc[j][r];
    }
}

// ---------------------------------------------------------------------------
// Small projections: gk = -exp(A_log)*softplus(x@Wgk + dt_bias),
//                    beta = sigmoid(x@Wb + bb).   One thread per (b,t,h).
// ---------------------------------------------------------------------------
__global__ __launch_bounds__(256) void proj_small_kernel(const float* __restrict__ x,
                                                         const float* __restrict__ Wgk,
                                                         const float* __restrict__ Wb,
                                                         const float* __restrict__ bb,
                                                         const float* __restrict__ A_log,
                                                         const float* __restrict__ dt_bias,
                                                         float* __restrict__ gkb,
                                                         float* __restrict__ betab) {
    int id = blockIdx.x * 256 + threadIdx.x;     // B*T*H
    if (id >= M_ * H_) return;
    int h  = id % H_;
    int bt = id / H_;
    const float* xr = x + (size_t)bt * C_;
    float s1 = 0.f, s2 = 0.f;
    for (int c = 0; c < C_; ++c) {
        float xv = xr[c];
        s1 += xv * Wgk[(size_t)c * H_ + h];
        s2 += xv * Wb [(size_t)c * H_ + h];
    }
    float z  = s1 + dt_bias[h];
    float sp = (z > 20.f) ? z : log1pf(__expf(z));
    gkb[id]   = -__expf(A_log[h]) * sp;
    betab[id] = sigf(s2 + bb[h]);
}

// ---------------------------------------------------------------------------
// Causal depthwise conv (K=4) + SiLU for q/k/v simultaneously.
// One thread per (b, t, channel).
// ---------------------------------------------------------------------------
__global__ __launch_bounds__(256) void conv_silu_kernel(const float* __restrict__ qlin,
                                                        const float* __restrict__ klin,
                                                        const float* __restrict__ vlin,
                                                        const float* __restrict__ wq,
                                                        const float* __restrict__ wk,
                                                        const float* __restrict__ wv,
                                                        float* __restrict__ qc,
                                                        float* __restrict__ kc,
                                                        float* __restrict__ vc) {
    int id = blockIdx.x * 256 + threadIdx.x;     // B*T*C
    if (id >= M_ * C_) return;
    int ch = id % C_;
    int bt = id / C_;
    int t  = bt % T_;
    float aq = 0.f, ak = 0.f, av = 0.f;
#pragma unroll
    for (int j = 0; j < CONVK; ++j) {
        int tt = t - (CONVK - 1) + j;
        if (tt >= 0) {
            size_t src = (size_t)id + (size_t)(tt - t) * C_;
            aq += wq[ch * CONVK + j] * qlin[src];
            ak += wk[ch * CONVK + j] * klin[src];
            av += wv[ch * CONVK + j] * vlin[src];
        }
    }
    qc[id] = siluf(aq);
    kc[id] = siluf(ak);
    vc[id] = siluf(av);
}

// ---------------------------------------------------------------------------
// Per-head L2 norm + RoPE (rot_dim = 32, half = 16), in place.
// One 64-thread block per (b, t, h).
// ---------------------------------------------------------------------------
__global__ __launch_bounds__(64) void l2norm_rope_kernel(float* __restrict__ data) {
    __shared__ float buf[DK_];
    int blk = blockIdx.x;                        // (b*T + t)*H + h
    int t   = (blk / H_) % T_;
    int d   = threadIdx.x;
    float* p = data + (size_t)blk * DK_;
    float v = p[d];
    buf[d] = v;
    __syncthreads();
    float ss = 1e-6f;
#pragma unroll
    for (int i = 0; i < DK_; ++i) ss += buf[i] * buf[i];
    float inv = rsqrtf(ss);
    if (d < 16) {
        float x1 = buf[d] * inv;
        float x2 = buf[d + 16] * inv;
        // inv_freq = 10000^(-d/16) = exp(-d * ln(10000)/16)
        float freq = __expf(-(float)d * (9.2103403719761836f / 16.f));
        float ang  = (float)t * freq;
        float cs = __cosf(ang), sn = __sinf(ang);
        p[d]      = x1 * cs - x2 * sn;
        p[d + 16] = x2 * cs + x1 * sn;
    } else if (d >= 32) {
        p[d] = buf[d] * inv;
    }
    // d in [16,32): already written by the d<16 threads
}

// ---------------------------------------------------------------------------
// Sequential gated delta-rule scan. One block per (b,h); 64 threads;
// thread j owns state column S[:, j] in registers.
// ---------------------------------------------------------------------------
__global__ __launch_bounds__(64) void scan_kernel(const float* __restrict__ qc,
                                                  const float* __restrict__ kc,
                                                  const float* __restrict__ vc,
                                                  const float* __restrict__ gkb,
                                                  const float* __restrict__ betab,
                                                  float* __restrict__ obuf) {
    const int bh = blockIdx.x;                   // b*H + h
    const int b  = bh / H_;
    const int h  = bh % H_;
    const int j  = threadIdx.x;
    __shared__ float kv[DK_], qv[DK_];
    float S[DK_];
#pragma unroll
    for (int i = 0; i < DK_; ++i) S[i] = 0.f;

    for (int t = 0; t < T_; ++t) {
        const size_t bt   = (size_t)b * T_ + t;
        const size_t base = bt * C_ + (size_t)h * DK_;
        kv[j] = kc[base + j];
        qv[j] = qc[base + j];
        __syncthreads();
        const float eg   = __expf(gkb[bt * H_ + h]);
        const float beta = betab[bt * H_ + h];
        const float vj   = vc[base + j];
        float pred = 0.f;
#pragma unroll
        for (int i = 0; i < DK_; ++i) pred += S[i] * kv[i];
        pred *= eg;
        const float delta = beta * (vj - pred);
        float o = 0.f;
#pragma unroll
        for (int i = 0; i < DK_; ++i) {
            S[i] = S[i] * eg + kv[i] * delta;
            o   += qv[i] * S[i];
        }
        obuf[base + j] = o;
        __syncthreads();
    }
}

// ---------------------------------------------------------------------------
// Per-head RMS norm * gnorm_w * silu(g), output bf16 (A operand of final GEMM).
// One 64-thread block per (b, t, h).
// ---------------------------------------------------------------------------
__global__ __launch_bounds__(64) void gate_out_kernel(const float* __restrict__ obuf,
                                                      const float* __restrict__ glin,
                                                      const float* __restrict__ gnw,
                                                      __bf16* __restrict__ og) {
    __shared__ float buf[DV_];
    int blk = blockIdx.x;                        // (b*T + t)*H + h
    int d   = threadIdx.x;
    const float* op = obuf + (size_t)blk * DV_;
    float v = op[d];
    buf[d] = v;
    __syncthreads();
    float ss = 0.f;
#pragma unroll
    for (int i = 0; i < DV_; ++i) ss += buf[i] * buf[i];
    float rms = rsqrtf(ss * (1.f / (float)DV_) + EPS_);
    float g   = glin[(size_t)blk * DV_ + d];
    og[(size_t)blk * DV_ + d] = (__bf16)(v * rms * gnw[d] * siluf(g));
}

// ---------------------------------------------------------------------------
// Host-side launcher
// ---------------------------------------------------------------------------
extern "C" void kernel_launch(void* const* d_in, const int* in_sizes, int n_in,
                              void* d_out, int out_size, void* d_ws, size_t ws_size,
                              hipStream_t stream) {
    const float* x    = (const float*)d_in[0];
    const float* Wq   = (const float*)d_in[1];
    const float* Wk   = (const float*)d_in[2];
    const float* Wv   = (const float*)d_in[3];
    const float* Wo   = (const float*)d_in[4];
    const float* Wg   = (const float*)d_in[5];
    const float* Wgk  = (const float*)d_in[6];
    const float* Wb   = (const float*)d_in[7];
    const float* bb   = (const float*)d_in[8];
    const float* A_log= (const float*)d_in[9];
    const float* cqw  = (const float*)d_in[10];
    const float* ckw  = (const float*)d_in[11];
    const float* cvw  = (const float*)d_in[12];
    const float* gnw  = (const float*)d_in[13];
    const float* dtb  = (const float*)d_in[14];

    // Workspace carve-up (256B aligned)
    char* base = (char*)d_ws;
    size_t off = 0;
    auto carve = [&](size_t bytes) -> char* {
        char* p = base + off;
        off = (off + bytes + 255) & ~(size_t)255;
        return p;
    };
    const size_t MC   = (size_t)M_ * C_;
    __bf16* xb   = (__bf16*)carve(MC * sizeof(__bf16));          // x in bf16
    __bf16* Wq_p = (__bf16*)carve((size_t)C_ * C_ * sizeof(__bf16));
    __bf16* Wk_p = (__bf16*)carve((size_t)C_ * C_ * sizeof(__bf16));
    __bf16* Wv_p = (__bf16*)carve((size_t)C_ * C_ * sizeof(__bf16));
    __bf16* Wg_p = (__bf16*)carve((size_t)C_ * C_ * sizeof(__bf16));
    __bf16* Wo_p = (__bf16*)carve((size_t)C_ * C_ * sizeof(__bf16));
    float* qlin = (float*)carve(MC * sizeof(float));
    float* klin = (float*)carve(MC * sizeof(float));
    float* vlin = (float*)carve(MC * sizeof(float));
    float* glin = (float*)carve(MC * sizeof(float));
    float* qc   = (float*)carve(MC * sizeof(float));
    float* kc   = (float*)carve(MC * sizeof(float));
    float* vc   = (float*)carve(MC * sizeof(float));
    float* gkb  = (float*)carve((size_t)M_ * H_ * sizeof(float));
    float* betab= (float*)carve((size_t)M_ * H_ * sizeof(float));
    // Buffer reuse: scan output overwrites qlin; gated bf16 output overwrites klin
    float*  obuf = qlin;
    __bf16* og   = (__bf16*)klin;

    // 1) x -> bf16
    cvt_bf16_kernel<<<(int)((MC + 255) / 256), 256, 0, stream>>>(x, xb, (int)MC);

    // 2) pack weights into WMMA B layout
    const int pk_blocks = (int)(((size_t)C_ * C_ + 255) / 256);
    pack_weight_kernel<<<pk_blocks, 256, 0, stream>>>(Wq, Wq_p, C_, C_);
    pack_weight_kernel<<<pk_blocks, 256, 0, stream>>>(Wk, Wk_p, C_, C_);
    pack_weight_kernel<<<pk_blocks, 256, 0, stream>>>(Wv, Wv_p, C_, C_);
    pack_weight_kernel<<<pk_blocks, 256, 0, stream>>>(Wg, Wg_p, C_, C_);
    pack_weight_kernel<<<pk_blocks, 256, 0, stream>>>(Wo, Wo_p, C_, C_);

    // 3) big projections via WMMA GEMM (TDM-staged B panels)
    dim3 gg(C_ / 64, M_ / 128);
    gemm_bf16_wmma<<<gg, 256, 0, stream>>>(xb, Wq_p, qlin, M_, C_, C_);
    gemm_bf16_wmma<<<gg, 256, 0, stream>>>(xb, Wk_p, klin, M_, C_, C_);
    gemm_bf16_wmma<<<gg, 256, 0, stream>>>(xb, Wv_p, vlin, M_, C_, C_);
    gemm_bf16_wmma<<<gg, 256, 0, stream>>>(xb, Wg_p, glin, M_, C_, C_);

    // 4) gk / beta small projections (f32)
    proj_small_kernel<<<(M_ * H_ + 255) / 256, 256, 0, stream>>>(
        x, Wgk, Wb, bb, A_log, dtb, gkb, betab);

    // 5) causal depthwise conv + SiLU
    conv_silu_kernel<<<(int)((MC + 255) / 256), 256, 0, stream>>>(
        qlin, klin, vlin, cqw, ckw, cvw, qc, kc, vc);

    // 6) l2norm + RoPE on q and k (in place)
    l2norm_rope_kernel<<<M_ * H_, 64, 0, stream>>>(qc);
    l2norm_rope_kernel<<<M_ * H_, 64, 0, stream>>>(kc);

    // 7) sequential gated delta-rule scan (obuf == qlin reuse)
    scan_kernel<<<B_ * H_, 64, 0, stream>>>(qc, kc, vc, gkb, betab, obuf);

    // 8) RMS norm + gate -> bf16 activations (og == klin reuse)
    gate_out_kernel<<<M_ * H_, 64, 0, stream>>>(obuf, glin, gnw, og);

    // 9) final projection: out = og @ Wo
    gemm_bf16_wmma<<<gg, 256, 0, stream>>>(og, Wo_p, (float*)d_out, M_, C_, C_);

    (void)in_sizes; (void)n_in; (void)out_size; (void)ws_size;
}
